// MultichannelMultiheadAttention_23287312679633
// MI455X (gfx1250) — compile-verified, matrix operand order checked by reading, not run on hardware
//
#include <hip/hip_runtime.h>
#include <math.h>

// ---------------- types ----------------
typedef __bf16 bf16;
typedef __attribute__((ext_vector_type(16))) __bf16 v16bf;
typedef __attribute__((ext_vector_type(8)))  __bf16 v8bf;
typedef __attribute__((ext_vector_type(8)))  float  v8f;

#define BB 4
#define CC 8
#define FF 512
#define WW 512
#define NH 8
#define HD 64

__device__ __forceinline__ bf16 f2bf(float f) {
    union { float f; unsigned u; } in; in.f = f;
    unsigned r = (in.u + 0x7FFFu + ((in.u >> 16) & 1u)) >> 16;
    union { unsigned short s; bf16 b; } out; out.s = (unsigned short)r;
    return out.b;
}

// Load a 16x32 bf16 WMMA A/B fragment from a row-major (ld-stride) matrix where
// the contraction dim K is contiguous. Per CDNA5 ISA layout for 16-bit A 16x32:
// lane L (m = L&15, hl = L>>4) holds K = e + (e>=8?8:0) + hl*8 for element e.
// -> two contiguous 8-element (16B) chunks at k0+hl*8 and k0+16+hl*8.
__device__ __forceinline__ v16bf load_frag(const bf16* base, int row, int ld,
                                           int k0, int lane) {
    int hl = (lane >> 4) & 1;
    const bf16* p0 = base + (size_t)row * ld + k0 + hl * 8;
    v8bf lo = *(const v8bf*)(p0);
    v8bf hi = *(const v8bf*)(p0 + 16);
    v16bf f;
#pragma unroll
    for (int e = 0; e < 8; ++e) { f[e] = lo[e]; f[e + 8] = hi[e]; }
    return f;
}

#define WMMA_BF16(A, B, C) \
    __builtin_amdgcn_wmma_f32_16x16x32_bf16(false, (A), false, (B), (short)0, (C), false, false)

// ---------------- 1) depthwise 9x9 conv, writes transposed bf16 out[w][h] ----
__global__ void dwconv9_kernel(const float* __restrict__ x,
                               const float* __restrict__ wgt,
                               bf16* __restrict__ outT) {
    __shared__ float tile[24][25];
    const int bc = blockIdx.z;
    const int c  = bc % CC;
    const int h0 = blockIdx.x * 16 - 4;
    const int w0 = blockIdx.y * 16 - 4;
    const float* xp = x + (size_t)bc * FF * WW;
    for (int idx = threadIdx.y * 16 + threadIdx.x; idx < 24 * 24; idx += 256) {
        int r = idx / 24, q = idx % 24;
        int hh = h0 + r, ww = w0 + q;
        float v = 0.f;
        if (hh >= 0 && hh < FF && ww >= 0 && ww < WW) v = xp[(size_t)hh * WW + ww];
        tile[r][q] = v;
    }
    __syncthreads();
    const int th = threadIdx.x, tw = threadIdx.y;
    const float* wk = wgt + c * 81;
    float s = 0.f;
#pragma unroll
    for (int i = 0; i < 9; ++i)
#pragma unroll
        for (int j = 0; j < 9; ++j)
            s += tile[th + i][tw + j] * wk[i * 9 + j];
    const int h = blockIdx.x * 16 + th;
    const int w = blockIdx.y * 16 + tw;
    outT[(size_t)bc * FF * WW + (size_t)w * FF + h] = f2bf(s);
}

// ---------------- 2) f32 -> bf16 convert ----------------
__global__ void cvt_bf16_kernel(const float* __restrict__ in,
                                bf16* __restrict__ out, int n) {
    int i = blockIdx.x * 256 + threadIdx.x;
    if (i < n) out[i] = f2bf(in[i]);
}

// ---------------- 3) generic 512^3 bf16 WMMA GEMM: C = A * Bt^T -------------
// 2x2 register blocking: each wave computes a 32x32 output tile (4 WMMAs per
// 2 A + 2 B fragment loads). 256 tiles/matrix, 8 waves/block -> grid(32, mats).
__global__ void gemm512_kernel(const bf16* __restrict__ Abase,
                               const bf16* __restrict__ Btbase,
                               float* __restrict__ Cbase, int a_mod) {
    const int L    = threadIdx.x & 31;
    const int wave = threadIdx.x >> 5;
    const int z    = blockIdx.y;
    const int tile = blockIdx.x * 8 + wave;   // 0..255
    const int tm   = (tile >> 4) * 32;
    const int tn   = (tile & 15) * 32;
    const bf16* A  = Abase  + (size_t)(z % a_mod) * FF * FF;
    const bf16* Bt = Btbase + (size_t)z * FF * FF;
    float*      Cm = Cbase  + (size_t)z * FF * FF;
    const int m = L & 15;
    v8f acc00 = {}, acc01 = {}, acc10 = {}, acc11 = {};
    for (int k0 = 0; k0 < 512; k0 += 32) {
        v16bf a0 = load_frag(A,  tm +      m, 512, k0, L);
        v16bf a1 = load_frag(A,  tm + 16 + m, 512, k0, L);
        v16bf b0 = load_frag(Bt, tn +      m, 512, k0, L);
        v16bf b1 = load_frag(Bt, tn + 16 + m, 512, k0, L);
        acc00 = WMMA_BF16(a0, b0, acc00);
        acc01 = WMMA_BF16(a0, b1, acc01);
        acc10 = WMMA_BF16(a1, b0, acc10);
        acc11 = WMMA_BF16(a1, b1, acc11);
    }
    const int mo = (L >> 4) * 8;
#pragma unroll
    for (int r = 0; r < 8; ++r) {
        float* row0 = Cm + (size_t)(tm + mo + r) * 512;
        float* row1 = Cm + (size_t)(tm + 16 + mo + r) * 512;
        row0[tn + m]      = acc00[r];
        row0[tn + 16 + m] = acc01[r];
        row1[tn + m]      = acc10[r];
        row1[tn + 16 + m] = acc11[r];
    }
}

// ---------------- 4) channel-mix + rotary + per-head pack (q, k) ------------
// out[b,d,n][i][hd] bf16, one thread per (b,d,n,i,pair)
__global__ void mix_rotary_pack_kernel(const float* __restrict__ Y,
                                       const float* __restrict__ dw,
                                       bf16* __restrict__ outp) {
    size_t idx = (size_t)blockIdx.x * 256 + threadIdx.x;
    int p = idx & 31;  size_t t = idx >> 5;
    int i = t & 511;   t >>= 9;
    int n = t & 7;     t >>= 3;
    int d = t & 7;     t >>= 3;
    int b = (int)t;
    if (b >= BB) return;
    const int g = n * HD + 2 * p;
    float z0 = 0.f, z1 = 0.f;
#pragma unroll
    for (int c = 0; c < CC; ++c) {
        float wdc = dw[d * CC + c];
        const float* yp = Y + ((size_t)(b * CC + c) * FF + g) * WW + i;
        z0 += wdc * yp[0];
        z1 += wdc * yp[WW];  // row g+1
    }
    float theta = (float)i * powf(10000.f, -(float)(2 * p) / (float)HD);
    float cs = cosf(theta), sn = sinf(theta);
    float o0 = z0 * cs - z1 * sn;
    float o1 = z1 * cs + z0 * sn;
    bf16* op = outp + ((((size_t)b * CC + d) * NH + n) * WW + i) * HD + 2 * p;
    op[0] = f2bf(o0);
    op[1] = f2bf(o1);
}

// ---------------- 5) channel-mix + transposed pack (v): vT[b,d,n][hd][i] ----
__global__ void mix_pack_v_kernel(const float* __restrict__ Y,
                                  const float* __restrict__ dw,
                                  bf16* __restrict__ vT) {
    size_t idx = (size_t)blockIdx.x * 256 + threadIdx.x;
    int i  = idx & 511; size_t t = idx >> 9;
    int hd = t & 63;    t >>= 6;
    int n  = t & 7;     t >>= 3;
    int d  = t & 7;     t >>= 3;
    int b  = (int)t;
    if (b >= BB) return;
    const int g = n * HD + hd;
    float z = 0.f;
#pragma unroll
    for (int c = 0; c < CC; ++c)
        z += dw[d * CC + c] * Y[((size_t)(b * CC + c) * FF + g) * WW + i];
    vT[((((size_t)b * CC + d) * NH + n) * HD + hd) * WW + i] = f2bf(z);
}

// ---------------- 6) qk = q*k^T*scale + prev, written to d_out --------------
// 2x2 blocking, K = HD = 64 (two WMMA k-steps per accumulator).
__global__ void qk_kernel(const bf16* __restrict__ qh,
                          const bf16* __restrict__ kh,
                          const float* __restrict__ prev,
                          float* __restrict__ qkout, float scale) {
    const int L    = threadIdx.x & 31;
    const int wave = threadIdx.x >> 5;
    const int z    = blockIdx.y;              // (b*C + c)*NH + n
    const int tile = blockIdx.x * 8 + wave;   // 0..255
    const int tm   = (tile >> 4) * 32;
    const int tn   = (tile & 15) * 32;
    const bf16* Q = qh + (size_t)z * WW * HD;
    const bf16* K = kh + (size_t)z * WW * HD;
    const int m = L & 15;
    v8f acc00 = {}, acc01 = {}, acc10 = {}, acc11 = {};
#pragma unroll
    for (int k0 = 0; k0 < HD; k0 += 32) {
        v16bf a0 = load_frag(Q, tm +      m, HD, k0, L);
        v16bf a1 = load_frag(Q, tm + 16 + m, HD, k0, L);
        v16bf b0 = load_frag(K, tn +      m, HD, k0, L);
        v16bf b1 = load_frag(K, tn + 16 + m, HD, k0, L);
        acc00 = WMMA_BF16(a0, b0, acc00);
        acc01 = WMMA_BF16(a0, b1, acc01);
        acc10 = WMMA_BF16(a1, b0, acc10);
        acc11 = WMMA_BF16(a1, b1, acc11);
    }
    const int mo = (L >> 4) * 8;
    const size_t zb = (size_t)z * WW * WW;
#pragma unroll
    for (int r = 0; r < 8; ++r) {
        size_t o0 = zb + (size_t)(tm + mo + r) * WW;
        size_t o1 = zb + (size_t)(tm + 16 + mo + r) * WW;
        qkout[o0 + tn + m]      = acc00[r] * scale + prev[o0 + tn + m];
        qkout[o0 + tn + 16 + m] = acc01[r] * scale + prev[o0 + tn + 16 + m];
        qkout[o1 + tn + m]      = acc10[r] * scale + prev[o1 + tn + m];
        qkout[o1 + tn + 16 + m] = acc11[r] * scale + prev[o1 + tn + 16 + m];
    }
}

// ---------------- 7) fused softmax + P*V (one wave per 16-row tile) ---------
__global__ void attn_kernel(const float* __restrict__ qk,
                            const bf16* __restrict__ vT,
                            bf16* __restrict__ aT) {
    __shared__ __align__(16) bf16 eS[16 * 512];
    __shared__ float rinv[16];
    const int L  = threadIdx.x;
    const int z  = blockIdx.y;
    const int i0 = blockIdx.x * 16;
    const float* qrow = qk + ((size_t)z * WW + i0) * WW;
    for (int r = 0; r < 16; ++r) {
        const float* row = qrow + (size_t)r * WW;
        float mx = -3.402823466e38f;
        for (int j = L; j < 512; j += 32) mx = fmaxf(mx, row[j]);
#pragma unroll
        for (int off = 16; off; off >>= 1) mx = fmaxf(mx, __shfl_xor(mx, off, 32));
        float s = 0.f;
        for (int j = L; j < 512; j += 32) {
            float e = expf(row[j] - mx);
            s += e;
            eS[r * 512 + j] = f2bf(e);
        }
#pragma unroll
        for (int off = 16; off; off >>= 1) s += __shfl_xor(s, off, 32);
        if (L == 0) rinv[r] = 1.f / s;
    }
    __syncthreads();
    const bf16* vb = vT + (size_t)z * HD * WW;
    const int m = L & 15;
    v8f acc0 = {}, acc1 = {}, acc2 = {}, acc3 = {};
    for (int k0 = 0; k0 < 512; k0 += 32) {
        v16bf pf = load_frag(eS, m, 512, k0, L);
        v16bf b0 = load_frag(vb,  0 + m, 512, k0, L);
        v16bf b1 = load_frag(vb, 16 + m, 512, k0, L);
        v16bf b2 = load_frag(vb, 32 + m, 512, k0, L);
        v16bf b3 = load_frag(vb, 48 + m, 512, k0, L);
        acc0 = WMMA_BF16(pf, b0, acc0);
        acc1 = WMMA_BF16(pf, b1, acc1);
        acc2 = WMMA_BF16(pf, b2, acc2);
        acc3 = WMMA_BF16(pf, b3, acc3);
    }
    const int bc = z >> 3, nh = z & 7;
    bf16* ab = aT + (size_t)bc * WW * FF;
    const int mo = (L >> 4) * 8;
#pragma unroll
    for (int r = 0; r < 8; ++r) {
        const int ii = i0 + mo + r;
        const float ri = rinv[mo + r];
        bf16* op = ab + (size_t)ii * FF + nh * HD + m;
        op[0]  = f2bf(acc0[r] * ri);
        op[16] = f2bf(acc1[r] * ri);
        op[32] = f2bf(acc2[r] * ri);
        op[48] = f2bf(acc3[r] * ri);
    }
}

// ---------------- 8) final channel mix -> d_out -----------------------------
__global__ void mix_out_kernel(const float* __restrict__ Y,
                               const float* __restrict__ dw,
                               float* __restrict__ out) {
    size_t idx = (size_t)blockIdx.x * 256 + threadIdx.x;
    int w = idx & 511; size_t t = idx >> 9;
    int g = t & 511;   t >>= 9;
    int d = t & 7;     t >>= 3;
    int b = (int)t;
    if (b >= BB) return;
    float z = 0.f;
#pragma unroll
    for (int c = 0; c < CC; ++c)
        z += dw[d * CC + c] * Y[((size_t)(b * CC + c) * FF + g) * WW + w];
    out[((size_t)(b * CC + d) * FF + g) * WW + w] = z;
}

// ---------------- host launcher ----------------
extern "C" void kernel_launch(void* const* d_in, const int* in_sizes, int n_in,
                              void* d_out, int out_size, void* d_ws, size_t ws_size,
                              hipStream_t stream) {
    const float* x    = (const float*)d_in[0];
    const float* prev = (const float*)d_in[1];
    const float* convw[3] = { (const float*)d_in[2], (const float*)d_in[3],
                              (const float*)d_in[4] };                // q,k,v conv
    const float* pw[4] = { (const float*)d_in[5], (const float*)d_in[7],
                           (const float*)d_in[9], (const float*)d_in[11] }; // q,k,v,o
    const float* dw[4] = { (const float*)d_in[6], (const float*)d_in[8],
                           (const float*)d_in[10], (const float*)d_in[12] };

    float* out   = (float*)d_out;
    float* qkout = out + (size_t)BB * CC * FF * WW;  // second output

    constexpr size_t PLANE  = (size_t)FF * WW;         // 262144
    constexpr size_t NBCF   = (size_t)BB * CC * PLANE; // 8,388,608 elems

    char* ws = (char*)d_ws;
    size_t off = 0;
    auto alloc = [&](size_t bytes) -> void* {
        void* p = ws + off;
        off += (bytes + 255) & ~(size_t)255;
        return p;
    };
    float* Y       = (float*)alloc(NBCF * 4);       // 32 MB, reused q/k/v/o
    bf16* convT[3];
    for (int t = 0; t < 3; ++t) convT[t] = (bf16*)alloc(NBCF * 2);
    bf16* pwb[4];
    for (int t = 0; t < 4; ++t) pwb[t] = (bf16*)alloc((size_t)CC * FF * FF * 2);
    bf16* qh  = (bf16*)alloc(NBCF * 2);
    bf16* kh  = (bf16*)alloc(NBCF * 2);
    bf16* vTb = (bf16*)alloc(NBCF * 2);
    bf16* aT  = (bf16*)alloc(NBCF * 2);

    // 1) weight conversion (4 pw tensors)
    {
        int n = CC * FF * FF;
        dim3 g((n + 255) / 256), b(256);
        for (int t = 0; t < 4; ++t)
            cvt_bf16_kernel<<<g, b, 0, stream>>>(pw[t], pwb[t], n);
    }
    // 2) depthwise convs (transposed bf16 outputs)
    {
        dim3 g(FF / 16, WW / 16, BB * CC), b(16, 16);
        for (int t = 0; t < 3; ++t)
            dwconv9_kernel<<<g, b, 0, stream>>>(x, convw[t], convT[t]);
    }

    const dim3 ggemm(32, BB * CC), bgemm(256);   // 256 (32x32) tiles / 8 waves
    const dim3 gpackqk((BB * CC * NH * WW * 32 + 255) / 256), b256(256);
    const dim3 gpackv((NBCF + 255) / 256);

    // 3) q: GEMM -> mix+rotary pack
    gemm512_kernel<<<ggemm, bgemm, 0, stream>>>(pwb[0], convT[0], Y, CC);
    mix_rotary_pack_kernel<<<gpackqk, b256, 0, stream>>>(Y, dw[0], qh);
    // 4) k
    gemm512_kernel<<<ggemm, bgemm, 0, stream>>>(pwb[1], convT[1], Y, CC);
    mix_rotary_pack_kernel<<<gpackqk, b256, 0, stream>>>(Y, dw[1], kh);
    // 5) v
    gemm512_kernel<<<ggemm, bgemm, 0, stream>>>(pwb[2], convT[2], Y, CC);
    mix_pack_v_kernel<<<gpackv, b256, 0, stream>>>(Y, dw[2], vTb);

    // 6) qk + prev -> d_out (second output)
    {
        dim3 g(32, BB * CC * NH), b(256);
        float scale = 1.0f / sqrtf((float)FF);
        qk_kernel<<<g, b, 0, stream>>>(qh, kh, prev, qkout, scale);
    }
    // 7) softmax + P*V
    {
        dim3 g(WW / 16, BB * CC * NH), b(32);
        attn_kernel<<<g, b, 0, stream>>>(qkout, vTb, aT);
    }
    // 8) output projection GEMM + channel mix -> d_out (first output)
    gemm512_kernel<<<ggemm, bgemm, 0, stream>>>(pwb[3], aT, Y, CC);
    mix_out_kernel<<<gpackv, b256, 0, stream>>>(Y, dw[3], out);
}
